// PCT_Heatmap_Tokenizer_9002251452924
// MI455X (gfx1250) — compile-verified, matrix operand order checked by reading, not run on hardware
//
#include <hip/hip_runtime.h>
#include <hip/hip_bf16.h>
#include <math.h>

// ---------------------------------------------------------------------------
// Shapes
// ---------------------------------------------------------------------------
#define BB        256
#define KK        17
#define AREA      3072
#define ENC_H     512
#define ENC_IH    512
#define ENC_TI    64
#define ENC_N     4
#define DEC_H     32
#define DEC_IH    64
#define DEC_N     1
#define TOK_N     34
#define TOK_CLS   2048
#define TOK_D     512
#define MROWS     (BB * KK)        // 4352  (= 34 * 128)
#define TROWS     (BB * TOK_N)     // 8704  (= 68 * 128)
#define REC_ELEMS (BB * KK * AREA) // 13369344

typedef __attribute__((ext_vector_type(16))) __bf16          v16bf;
typedef __attribute__((ext_vector_type(16))) unsigned short  v16us;
typedef __attribute__((ext_vector_type(8)))  unsigned short  v8us;
typedef __attribute__((ext_vector_type(8)))  float           v8f;

__device__ __forceinline__ unsigned short f2bf(float f) {
    unsigned int u = __float_as_uint(f);
    return (unsigned short)((u + 0x7FFFu + ((u >> 16) & 1u)) >> 16);
}
__device__ __forceinline__ float gelu_exact(float x) {
    return 0.5f * x * (1.0f + erff(x * 0.70710678118654752440f));
}
__device__ __forceinline__ v8us pack8(const float4 f0, const float4 f1) {
    v8us c;
    c[0] = f2bf(f0.x); c[1] = f2bf(f0.y); c[2] = f2bf(f0.z); c[3] = f2bf(f0.w);
    c[4] = f2bf(f1.x); c[5] = f2bf(f1.y); c[6] = f2bf(f1.z); c[7] = f2bf(f1.w);
    return c;
}

// ---------------------------------------------------------------------------
// One-time weight packing:  W (f32, [K,N] row-major, or [N,K] if trans) ->
// bf16 WMMA B-fragment layout, N zero-padded to Npad (multiple of 16/64).
// Fragment for (ntile, ktile): 32 lanes x 16 contiguous bf16 (32B per lane).
// Linear order: [ntile][ktile][lane].
// ---------------------------------------------------------------------------
__global__ __launch_bounds__(256)
void k_pack_w(const float* __restrict__ W, unsigned short* __restrict__ Wp,
              int Kd, int N, int Npad, int trans)
{
    const int kt_cnt = Kd >> 5;
    const int total  = (Npad >> 4) * kt_cnt * 32;
    const int t = blockIdx.x * 256 + threadIdx.x;
    if (t >= total) return;
    const int lane  = t & 31;
    const int kt    = (t >> 5) % kt_cnt;
    const int ntile = (t >> 5) / kt_cnt;
    const int n     = ntile * 16 + (lane & 15);
    const int kh    = (lane >> 4) << 3;
    v16us c;
    #pragma unroll
    for (int e = 0; e < 8; ++e) {
        int k0 = kt * 32 + kh + e;
        int k1 = kt * 32 + 16 + kh + e;
        float v0 = 0.0f, v1 = 0.0f;
        if (n < N) {
            v0 = trans ? W[(size_t)n * Kd + k0] : W[(size_t)k0 * N + n];
            v1 = trans ? W[(size_t)n * Kd + k1] : W[(size_t)k1 * N + n];
        }
        c[e]     = f2bf(v0);
        c[e + 8] = f2bf(v1);
    }
    ((v16us*)Wp)[((size_t)ntile * kt_cnt + kt) * 32 + lane] = c;
}

// ---------------------------------------------------------------------------
// GEMM: C[M,N] = epilogue(A[M,K] @ B + bias) (+ residual)
// A: f32 row-major, staged through LDS into fragment layout.
// Bp: pre-packed bf16 fragments (Npad-padded), streamed from global/L2.
// Block = 128 threads (4 waves) computes 128x64; wave = 32 rows x 64 cols,
// 8 WMMAs per 32-wide K-step. REQUIRES M%128==0, K%32==0 (true below).
// act: 0 = none, 1 = exact GELU
// ---------------------------------------------------------------------------
__global__ __launch_bounds__(128)
void k_gemm_bf16(const float* __restrict__ A, const unsigned short* __restrict__ Bp,
                 const float* __restrict__ bias, const float* __restrict__ resid,
                 float* __restrict__ C, int M, int N, int Kd, int act)
{
    __shared__ v16us lAp[8][32];   // 8 row-tiles of 16 x fragment lane : 8 KB

    const int tid  = threadIdx.x;
    const int wave = tid >> 5;
    const int lane = tid & 31;
    const int m0   = blockIdx.y * 128;
    const int n0   = blockIdx.x * 64;
    const int kt_cnt = Kd >> 5;
    const v16us* Bp16 = (const v16us*)Bp;

    v8f acc0[4] = {}, acc1[4] = {};

    for (int kb = 0; kb < Kd; kb += 32) {
        // ---- stage A tile 128x32 into fragment layout: 4 chunks/thread ----
        #pragma unroll
        for (int i = 0; i < 4; ++i) {
            int chunk = tid + i * 128;
            int r  = chunk >> 2;        // row in tile 0..127
            int kg = chunk & 3;         // K-group 0..3
            const float4* ap =
                (const float4*)(A + (size_t)(m0 + r) * Kd + kb + kg * 8);
            v8us c = pack8(ap[0], ap[1]);
            ((v8us*)&lAp[r >> 4][(r & 15) + 16 * (kg & 1)])[kg >> 1] = c;
        }
        __syncthreads();

        const int kt = kb >> 5;
        v16bf a0 = __builtin_bit_cast(v16bf, lAp[wave * 2 + 0][lane]);
        v16bf a1 = __builtin_bit_cast(v16bf, lAp[wave * 2 + 1][lane]);
        #pragma unroll
        for (int nt = 0; nt < 4; ++nt) {
            v16us bu = Bp16[((size_t)((n0 >> 4) + nt) * kt_cnt + kt) * 32 + lane];
            v16bf b  = __builtin_bit_cast(v16bf, bu);
            acc0[nt] = __builtin_amdgcn_wmma_f32_16x16x32_bf16(
                false, a0, false, b, (short)0, acc0[nt], false, false);
            acc1[nt] = __builtin_amdgcn_wmma_f32_16x16x32_bf16(
                false, a1, false, b, (short)0, acc1[nt], false, false);
        }
        __syncthreads();
    }

    // epilogue: lane holds col n = (lane&15); row = e + 8*(lane>=16)
    const int l    = lane & 15;
    const int rsel = (lane >> 4) << 3;
    #pragma unroll
    for (int nt = 0; nt < 4; ++nt) {
        int gn = n0 + nt * 16 + l;
        if (gn >= N) continue;
        float bv = bias ? bias[gn] : 0.0f;
        #pragma unroll
        for (int e = 0; e < 8; ++e) {
            int gm = m0 + wave * 32 + rsel + e;
            float v = acc0[nt][e] + bv;
            if (act == 1) v = gelu_exact(v);
            if (resid)    v += resid[(size_t)gm * N + gn];
            C[(size_t)gm * N + gn] = v;
            int gm2 = gm + 16;
            float v2 = acc1[nt][e] + bv;
            if (act == 1) v2 = gelu_exact(v2);
            if (resid)    v2 += resid[(size_t)gm2 * N + gn];
            C[(size_t)gm2 * N + gn] = v2;
        }
    }
}

// ---------------------------------------------------------------------------
// Fused VQ argmin.  Block = 128 threads (4 waves); each wave owns a 16-row
// f tile, fragment-packed once into LDS.  The codebook is pre-packed bf16
// B-fragments in global (fits in L2): one 32B load per lane per K-step,
// 16 back-to-back WMMAs per 16-code tile, running argmin in registers.
// ---------------------------------------------------------------------------
__global__ __launch_bounds__(128)
void k_vq_argmin(const float* __restrict__ f, const unsigned short* __restrict__ cbp,
                 const float* __restrict__ cbn, int* __restrict__ idx_ws,
                 int* __restrict__ idx_out)
{
    __shared__ v16us lFp[4][16][32];   // [wave][kstep][fragment lane] : 64 KB

    const int tid  = threadIdx.x;
    const int wave = tid >> 5;
    const int lane = tid & 31;
    const int row0 = (blockIdx.x * 4 + wave) * 16;   // grid sized exactly
    const v16us* cbp16 = (const v16us*)cbp;

    // ---- pack this wave's 16x512 f tile: 1024 chunks of 8; 32/lane ----
    #pragma unroll 4
    for (int i = 0; i < 32; ++i) {
        int chunk = lane + i * 32;
        int r     = chunk >> 6;         // 0..15
        int rem   = chunk & 63;
        int kstep = rem >> 2;           // 0..15
        int kg    = rem & 3;
        const float4* fp =
            (const float4*)(f + (size_t)(row0 + r) * TOK_D + kstep * 32 + kg * 8);
        v8us c = pack8(fp[0], fp[1]);
        ((v8us*)&lFp[wave][kstep][r + 16 * (kg & 1)])[kg >> 1] = c;
    }
    __syncthreads();

    const int l = lane & 15;
    float best[8];
    int   bidx[8];
    #pragma unroll
    for (int e = 0; e < 8; ++e) { best[e] = 3.4e38f; bidx[e] = 0; }

    for (int ntile = 0; ntile < (TOK_CLS >> 4); ++ntile) {
        const size_t base = (size_t)ntile * 16 * 32;   // 16 ktiles per ntile
        if (ntile + 1 < (TOK_CLS >> 4))
            __builtin_prefetch(&cbp16[base + 16 * 32 + lane], 0, 1);

        v8f acc = {};
        #pragma unroll
        for (int kstep = 0; kstep < 16; ++kstep) {
            v16bf a = __builtin_bit_cast(v16bf, lFp[wave][kstep][lane]);
            v16bf b = __builtin_bit_cast(v16bf, cbp16[base + kstep * 32 + lane]);
            acc = __builtin_amdgcn_wmma_f32_16x16x32_bf16(
                false, a, false, b, (short)0, acc, false, false);
        }
        const int   code = ntile * 16 + l;
        const float cv   = cbn[code];
        #pragma unroll
        for (int e = 0; e < 8; ++e) {
            float v = cv - 2.0f * acc[e];
            if (v < best[e]) { best[e] = v; bidx[e] = code; }
        }
    }

    // min-reduce over the 16 lanes sharing a row set (ties -> lowest index)
    #pragma unroll
    for (int e = 0; e < 8; ++e) {
        float bv = best[e];
        int   bi = bidx[e];
        for (int off = 1; off < 16; off <<= 1) {
            float ov = __shfl_xor(bv, off, 32);
            int   oi = __shfl_xor(bi, off, 32);
            if (ov < bv || (ov == bv && oi < bi)) { bv = ov; bi = oi; }
        }
        if (l == 0) {
            int gm = row0 + ((lane >> 4) << 3) + e;
            idx_ws[gm]  = bi;
            idx_out[gm] = bi;
        }
    }
}

// ---------------------------------------------------------------------------
// LayerNorm over last dim D (wave per row, wave32 shuffle reductions)
// ---------------------------------------------------------------------------
__global__ __launch_bounds__(128)
void k_layernorm(const float* __restrict__ x, const float* __restrict__ g,
                 const float* __restrict__ b, float* __restrict__ y, int M, int D)
{
    const int wave = threadIdx.x >> 5;
    const int lane = threadIdx.x & 31;
    const int row  = blockIdx.x * 4 + wave;
    if (row >= M) return;
    const float* xr = x + (size_t)row * D;
    float s = 0.0f, s2 = 0.0f;
    for (int c = lane; c < D; c += 32) { float v = xr[c]; s += v; s2 += v * v; }
    for (int off = 16; off >= 1; off >>= 1) {
        s  += __shfl_xor(s,  off, 32);
        s2 += __shfl_xor(s2, off, 32);
    }
    const float mean = s / (float)D;
    const float var  = s2 / (float)D - mean * mean;
    const float rstd = rsqrtf(var + 1e-5f);
    float* yr = y + (size_t)row * D;
    for (int c = lane; c < D; c += 32)
        yr[c] = (xr[c] - mean) * rstd * g[c] + b[c];
}

// ---------------------------------------------------------------------------
// Token-mixing MLP (fixed J=17, T=64), fused residual: u = x + mix(LN(x))
// ---------------------------------------------------------------------------
__global__ __launch_bounds__(256)
void k_token_mlp(const float* __restrict__ x, const float* __restrict__ y,
                 const float* __restrict__ tw1, const float* __restrict__ tb1,
                 const float* __restrict__ tw2, const float* __restrict__ tb2,
                 float* __restrict__ u, int H)
{
    const int t = blockIdx.x * blockDim.x + threadIdx.x;
    if (t >= BB * H) return;
    const int b = t / H, h = t % H;
    const float* yb = y + (size_t)b * KK * H + h;
    float yv[KK], o[KK];
    #pragma unroll
    for (int j = 0; j < KK; ++j) yv[j] = yb[(size_t)j * H];
    #pragma unroll
    for (int j = 0; j < KK; ++j) o[j] = tb2[j];
    for (int tt = 0; tt < ENC_TI; ++tt) {
        float hs = tb1[tt];
        #pragma unroll
        for (int j = 0; j < KK; ++j) hs += yv[j] * tw1[j * ENC_TI + tt];
        float hg = gelu_exact(hs);
        #pragma unroll
        for (int j = 0; j < KK; ++j) o[j] += hg * tw2[tt * KK + j];
    }
    const float* xb = x + (size_t)b * KK * H + h;
    float* ub = u + (size_t)b * KK * H + h;
    #pragma unroll
    for (int j = 0; j < KK; ++j) ub[(size_t)j * H] = xb[(size_t)j * H] + o[j];
}

// ---------------------------------------------------------------------------
// Joint<->token map: out[b,t,h] = bias[t] + sum_j in[b,j,h]*W[j,t]
// ---------------------------------------------------------------------------
__global__ __launch_bounds__(256)
void k_jointmap(const float* __restrict__ in, const float* __restrict__ W,
                const float* __restrict__ bias, float* __restrict__ out,
                int Jin, int Tout, int H)
{
    const int t = blockIdx.x * blockDim.x + threadIdx.x;
    if (t >= BB * H) return;
    const int b = t / H, h = t % H;
    const float* ib = in + (size_t)b * Jin * H + h;
    float vin[TOK_N];
    for (int j = 0; j < Jin; ++j) vin[j] = ib[(size_t)j * H];
    float* ob = out + (size_t)b * Tout * H + h;
    for (int tt = 0; tt < Tout; ++tt) {
        float s = bias[tt];
        for (int j = 0; j < Jin; ++j) s += vin[j] * W[j * Tout + tt];
        ob[(size_t)tt * H] = s;
    }
}

// ---------------------------------------------------------------------------
// misc small kernels
// ---------------------------------------------------------------------------
__global__ void k_blend(float* __restrict__ x, const float* __restrict__ tw,
                        const float* __restrict__ inv)
{
    const int i = blockIdx.x * blockDim.x + threadIdx.x;
    if (i >= MROWS * ENC_H) return;
    const float w = tw[i >> 9];
    x[i] = x[i] * w + inv[i & 511] * (1.0f - w);
}

__global__ void k_cbnorm(const float* __restrict__ cb, float* __restrict__ cbn)
{
    const int c = blockIdx.x * blockDim.x + threadIdx.x;
    if (c >= TOK_CLS) return;
    const float* r = cb + (size_t)c * TOK_D;
    float s = 0.0f;
    for (int k = 0; k < TOK_D; ++k) s += r[k] * r[k];
    cbn[c] = s;
}

__global__ void k_gather(float* __restrict__ part, const float* __restrict__ cb,
                         const int* __restrict__ idx)
{
    const int i = blockIdx.x * blockDim.x + threadIdx.x;
    if (i >= TROWS * TOK_D) return;
    part[i] = cb[(size_t)idx[i >> 9] * TOK_D + (i & 511)];
}

// ---------------------------------------------------------------------------
// host
// ---------------------------------------------------------------------------
extern "C" void kernel_launch(void* const* d_in, const int* in_sizes, int n_in,
                              void* d_out, int out_size, void* d_ws, size_t ws_size,
                              hipStream_t stream) {
    (void)in_sizes; (void)n_in; (void)out_size; (void)ws_size;

    const float* target   = (const float*)d_in[0];
    const float* tweight  = (const float*)d_in[1];
    // d_in[2] cls_logits unused
    const float* inv_tok  = (const float*)d_in[3];
    const float* se_w     = (const float*)d_in[4];
    const float* se_b     = (const float*)d_in[5];
    const float* e_ln1g   = (const float*)d_in[6];
    const float* e_ln1b   = (const float*)d_in[7];
    const float* e_tw1    = (const float*)d_in[8];
    const float* e_tb1    = (const float*)d_in[9];
    const float* e_tw2    = (const float*)d_in[10];
    const float* e_tb2    = (const float*)d_in[11];
    const float* e_ln2g   = (const float*)d_in[12];
    const float* e_ln2b   = (const float*)d_in[13];
    const float* e_cw1    = (const float*)d_in[14];
    const float* e_cb1    = (const float*)d_in[15];
    const float* e_cw2    = (const float*)d_in[16];
    const float* e_cb2    = (const float*)d_in[17];
    const float* e_lng    = (const float*)d_in[18];
    const float* e_lnb    = (const float*)d_in[19];
    const float* tok_w    = (const float*)d_in[20];
    const float* tok_b    = (const float*)d_in[21];
    const float* fe_w     = (const float*)d_in[22];
    const float* fe_b     = (const float*)d_in[23];
    const float* codebook = (const float*)d_in[24];
    const float* dtok_w   = (const float*)d_in[25];
    const float* dtok_b   = (const float*)d_in[26];
    const float* ds_w     = (const float*)d_in[27];
    const float* ds_b     = (const float*)d_in[28];
    const float* d_ln1g   = (const float*)d_in[29];
    const float* d_ln1b   = (const float*)d_in[30];
    const float* d_tw1    = (const float*)d_in[31];
    const float* d_tb1    = (const float*)d_in[32];
    const float* d_tw2    = (const float*)d_in[33];
    const float* d_tb2    = (const float*)d_in[34];
    const float* d_ln2g   = (const float*)d_in[35];
    const float* d_ln2b   = (const float*)d_in[36];
    const float* d_cw1    = (const float*)d_in[37];
    const float* d_cb1    = (const float*)d_in[38];
    const float* d_cw2    = (const float*)d_in[39];
    const float* d_cb2    = (const float*)d_in[40];
    const float* d_lng    = (const float*)d_in[41];
    const float* d_lnb    = (const float*)d_in[42];
    const float* re_w     = (const float*)d_in[43];
    const float* re_b     = (const float*)d_in[44];

    // workspace bump allocator (32B aligned)
    float* wsp = (float*)d_ws;
    size_t off = 0;
    auto alloc = [&](size_t n) {
        off = (off + 7) & ~(size_t)7;
        float* p = wsp + off; off += n; return p;
    };
    auto allocus = [&](size_t n_us) {
        off = (off + 7) & ~(size_t)7;
        unsigned short* p = (unsigned short*)(wsp + off);
        off += (n_us + 1) / 2; return p;
    };

    float* xa   = alloc((size_t)MROWS * ENC_H);
    float* xb   = alloc((size_t)MROWS * ENC_H);
    float* xu   = alloc((size_t)MROWS * ENC_H);
    float* xz   = alloc((size_t)MROWS * ENC_IH);
    float* t34  = alloc((size_t)TROWS * ENC_H);
    float* ff   = alloc((size_t)TROWS * TOK_D);
    float* part = alloc((size_t)TROWS * TOK_D);
    float* pj   = alloc((size_t)MROWS * TOK_D);
    float* dd   = alloc((size_t)MROWS * DEC_H);
    float* dln  = alloc((size_t)MROWS * DEC_H);
    float* du   = alloc((size_t)MROWS * DEC_H);
    float* dz   = alloc((size_t)MROWS * DEC_IH);
    float* cbn  = alloc(TOK_CLS);
    int*   idxw = (int*)alloc(TROWS);

    // packed bf16 weights (fragment layout, N padded to multiple of 64)
    unsigned short* p_sew  = allocus((size_t)AREA  * 512);        // 3072x512
    unsigned short* p_cw1[ENC_N], *p_cw2[ENC_N];
    for (int i = 0; i < ENC_N; ++i) {
        p_cw1[i] = allocus((size_t)ENC_H * ENC_IH);
        p_cw2[i] = allocus((size_t)ENC_IH * ENC_H);
    }
    unsigned short* p_few  = allocus((size_t)ENC_H * TOK_D);
    unsigned short* p_cb   = allocus((size_t)TOK_D * TOK_CLS);    // transposed
    unsigned short* p_dsw  = allocus((size_t)TOK_D * 64);         // N=32 ->64
    unsigned short* p_dcw1 = allocus((size_t)DEC_H * 64);         // N=64
    unsigned short* p_dcw2 = allocus((size_t)DEC_IH * 64);        // N=32 ->64
    unsigned short* p_rew  = allocus((size_t)DEC_H * AREA);

    float* rec     = (float*)d_out;
    int*   idx_out = (int*)d_out + REC_ELEMS;

    dim3 blk128(128), blk256(256);
    auto gemm_grid = [](int M, int N) { return dim3((N + 63) / 64, M / 128); };
    auto pack_grid = [](int Kd, int Npad) {
        return dim3(((Npad / 16) * (Kd / 32) * 32 + 255) / 256);
    };

    // ---- 0. pack all GEMM weights + codebook to bf16 fragments ----
    k_pack_w<<<pack_grid(AREA, 512), blk256, 0, stream>>>(se_w, p_sew, AREA, 512, 512, 0);
    for (int i = 0; i < ENC_N; ++i) {
        k_pack_w<<<pack_grid(ENC_H, ENC_IH), blk256, 0, stream>>>(
            e_cw1 + (size_t)i * ENC_H * ENC_IH, p_cw1[i], ENC_H, ENC_IH, ENC_IH, 0);
        k_pack_w<<<pack_grid(ENC_IH, ENC_H), blk256, 0, stream>>>(
            e_cw2 + (size_t)i * ENC_IH * ENC_H, p_cw2[i], ENC_IH, ENC_H, ENC_H, 0);
    }
    k_pack_w<<<pack_grid(ENC_H, TOK_D), blk256, 0, stream>>>(fe_w, p_few, ENC_H, TOK_D, TOK_D, 0);
    k_pack_w<<<pack_grid(TOK_D, TOK_CLS), blk256, 0, stream>>>(
        codebook, p_cb, TOK_D, TOK_CLS, TOK_CLS, 1);   // trans: cb is [N,K]
    k_pack_w<<<pack_grid(TOK_D, 64), blk256, 0, stream>>>(ds_w, p_dsw, TOK_D, DEC_H, 64, 0);
    k_pack_w<<<pack_grid(DEC_H, 64), blk256, 0, stream>>>(d_cw1, p_dcw1, DEC_H, DEC_IH, 64, 0);
    k_pack_w<<<pack_grid(DEC_IH, 64), blk256, 0, stream>>>(d_cw2, p_dcw2, DEC_IH, DEC_H, 64, 0);
    k_pack_w<<<pack_grid(DEC_H, AREA), blk256, 0, stream>>>(re_w, p_rew, DEC_H, AREA, AREA, 0);

    // ---- 1. encoder embedding + invisible-token blend ----
    k_gemm_bf16<<<gemm_grid(MROWS, ENC_H), blk128, 0, stream>>>(
        target, p_sew, se_b, nullptr, xa, MROWS, ENC_H, AREA, 0);
    k_blend<<<(MROWS * ENC_H + 255) / 256, blk256, 0, stream>>>(xa, tweight, inv_tok);

    // ---- 2. encoder mixer layers ----
    for (int i = 0; i < ENC_N; ++i) {
        k_layernorm<<<(MROWS + 3) / 4, blk128, 0, stream>>>(
            xa, e_ln1g + i * ENC_H, e_ln1b + i * ENC_H, xb, MROWS, ENC_H);
        k_token_mlp<<<(BB * ENC_H + 255) / 256, blk256, 0, stream>>>(
            xa, xb, e_tw1 + i * KK * ENC_TI, e_tb1 + i * ENC_TI,
            e_tw2 + i * ENC_TI * KK, e_tb2 + i * KK, xu, ENC_H);
        k_layernorm<<<(MROWS + 3) / 4, blk128, 0, stream>>>(
            xu, e_ln2g + i * ENC_H, e_ln2b + i * ENC_H, xb, MROWS, ENC_H);
        k_gemm_bf16<<<gemm_grid(MROWS, ENC_IH), blk128, 0, stream>>>(
            xb, p_cw1[i], e_cb1 + i * ENC_IH, nullptr, xz, MROWS, ENC_IH, ENC_H, 1);
        k_gemm_bf16<<<gemm_grid(MROWS, ENC_H), blk128, 0, stream>>>(
            xz, p_cw2[i], e_cb2 + i * ENC_H, xu, xa, MROWS, ENC_H, ENC_IH, 0);
    }

    // ---- 3. final encoder LN, joints->tokens ----
    k_layernorm<<<(MROWS + 3) / 4, blk128, 0, stream>>>(xa, e_lng, e_lnb, xb, MROWS, ENC_H);
    k_jointmap<<<(BB * ENC_H + 255) / 256, blk256, 0, stream>>>(
        xb, tok_w, tok_b, t34, KK, TOK_N, ENC_H);

    // ---- 4. feature embed + VQ ----
    k_gemm_bf16<<<gemm_grid(TROWS, TOK_D), blk128, 0, stream>>>(
        t34, p_few, fe_b, nullptr, ff, TROWS, TOK_D, ENC_H, 0);
    k_cbnorm<<<(TOK_CLS + 255) / 256, blk256, 0, stream>>>(codebook, cbn);
    k_vq_argmin<<<TROWS / 64, blk128, 0, stream>>>(ff, p_cb, cbn, idxw, idx_out);
    k_gather<<<((size_t)TROWS * TOK_D + 255) / 256, blk256, 0, stream>>>(part, codebook, idxw);

    // ---- 5. tokens->joints, decode projection ----
    k_jointmap<<<(BB * TOK_D + 255) / 256, blk256, 0, stream>>>(
        part, dtok_w, dtok_b, pj, TOK_N, KK, TOK_D);
    k_gemm_bf16<<<gemm_grid(MROWS, DEC_H), blk128, 0, stream>>>(
        pj, p_dsw, ds_b, nullptr, dd, MROWS, DEC_H, TOK_D, 0);

    // ---- 6. decoder mixer layer ----
    for (int i = 0; i < DEC_N; ++i) {
        k_layernorm<<<(MROWS + 3) / 4, blk128, 0, stream>>>(
            dd, d_ln1g + i * DEC_H, d_ln1b + i * DEC_H, dln, MROWS, DEC_H);
        k_token_mlp<<<(BB * DEC_H + 255) / 256, blk256, 0, stream>>>(
            dd, dln, d_tw1 + i * KK * ENC_TI, d_tb1 + i * ENC_TI,
            d_tw2 + i * ENC_TI * KK, d_tb2 + i * KK, du, DEC_H);
        k_layernorm<<<(MROWS + 3) / 4, blk128, 0, stream>>>(
            du, d_ln2g + i * DEC_H, d_ln2b + i * DEC_H, dln, MROWS, DEC_H);
        k_gemm_bf16<<<gemm_grid(MROWS, DEC_IH), blk128, 0, stream>>>(
            dln, p_dcw1, d_cb1 + i * DEC_IH, nullptr, dz, MROWS, DEC_IH, DEC_H, 1);
        k_gemm_bf16<<<gemm_grid(MROWS, DEC_H), blk128, 0, stream>>>(
            dz, p_dcw2, d_cb2 + i * DEC_H, du, dd, MROWS, DEC_H, DEC_IH, 0);
    }

    // ---- 7. final decoder LN + reconstruction into d_out ----
    k_layernorm<<<(MROWS + 3) / 4, blk128, 0, stream>>>(dd, d_lng, d_lnb, dln, MROWS, DEC_H);
    k_gemm_bf16<<<gemm_grid(MROWS, AREA), blk128, 0, stream>>>(
        dln, p_rew, re_b, nullptr, rec, MROWS, AREA, DEC_H, 0);
}
